// DeepseekV2MOE_49709951483962
// MI455X (gfx1250) — matrix operand order repaired and checked by reading, not compiled, not run on hardware
//
#include <hip/hip_runtime.h>
#include <hip/hip_bf16.h>
#include <math.h>

// ---------------------------------------------------------------------------
// DeepSeek-V2 MoE for MI455X (gfx1250): bf16 WMMA grouped GEMMs, wave32.
// M=32 tiles per wave-column (2 WMMA per B fragment) to halve B-load cost.
// ---------------------------------------------------------------------------

typedef __attribute__((ext_vector_type(16))) __bf16 v16bf;
typedef __attribute__((ext_vector_type(8)))  __bf16 v8bf;
typedef __attribute__((ext_vector_type(4)))  __bf16 v4bf;
typedef __attribute__((ext_vector_type(8)))  float  v8f;

#define T_TOK   1024
#define H_DIM   1024
#define E_EXP   64
#define I_DIM   512
#define K_TOP   6
#define G_GRP   8
#define TKG_SEL 3
#define CAP     256
#define ISH     1024      // I * N_SHARED
#define SCALE_F 2.5f

// ------------------------------ Router -------------------------------------
__global__ void moe_router_kernel(const float* __restrict__ x,
                                  const float* __restrict__ gate_w,
                                  const float* __restrict__ e_bias,
                                  int* __restrict__ eidx_flat,
                                  float* __restrict__ w_flat) {
    __shared__ float red[256];
    __shared__ float logit[E_EXP];
    const int t   = blockIdx.x;
    const int tid = threadIdx.x;
    const int e   = tid & (E_EXP - 1);
    const int p   = tid >> 6;                 // 0..3
    float sum = 0.f;
    const float* xr = x + (size_t)t * H_DIM + p * 256;
    const float* gw = gate_w + (size_t)(p * 256) * E_EXP + e;
    for (int h = 0; h < 256; ++h)
        sum += xr[h] * gw[(size_t)h * E_EXP];
    red[tid] = sum;
    __syncthreads();
    if (tid < E_EXP)
        logit[tid] = red[tid] + red[tid + 64] + red[tid + 128] + red[tid + 192];
    __syncthreads();
    if (tid == 0) {
        float sc[E_EXP], sb[E_EXP];
        for (int i = 0; i < E_EXP; ++i) {
            float s = 1.f / (1.f + expf(-logit[i]));
            sc[i] = s;
            sb[i] = s + e_bias[i];
        }
        float gs[G_GRP];
        for (int g = 0; g < G_GRP; ++g) {
            float m1 = -1e30f, m2 = -1e30f;
            for (int j = 0; j < 8; ++j) {
                float v = sb[g * 8 + j];
                if (v > m1) { m2 = m1; m1 = v; }
                else if (v > m2) { m2 = v; }
            }
            gs[g] = m1 + m2;
        }
        int gsel[G_GRP];
        for (int g = 0; g < G_GRP; ++g) gsel[g] = 0;
        for (int r = 0; r < TKG_SEL; ++r) {       // top-3 groups, ties -> low idx
            float best = -1e30f; int bi = 0;
            for (int g = 0; g < G_GRP; ++g)
                if (!gsel[g] && gs[g] > best) { best = gs[g]; bi = g; }
            gsel[bi] = 1;
        }
        int taken[E_EXP];
        for (int i = 0; i < E_EXP; ++i) taken[i] = 0;
        int sel[K_TOP];
        for (int r = 0; r < K_TOP; ++r) {         // top-6 experts within mask
            float best = -1e30f; int bi = 0;
            for (int i = 0; i < E_EXP; ++i)
                if (gsel[i >> 3] && !taken[i] && sb[i] > best) { best = sb[i]; bi = i; }
            taken[bi] = 1; sel[r] = bi;
        }
        float wsum = 0.f;
        for (int r = 0; r < K_TOP; ++r) wsum += sc[sel[r]];
        const float inv = 1.f / wsum;
        for (int r = 0; r < K_TOP; ++r) {
            eidx_flat[t * K_TOP + r] = sel[r];
            w_flat[t * K_TOP + r]    = sc[sel[r]] * inv;
        }
    }
}

// ------------------------------ Dispatch -----------------------------------
__global__ void moe_dispatch_kernel(const int* __restrict__ eidx_flat,
                                    int* __restrict__ pos_flat,
                                    int* __restrict__ tok_list,
                                    int* __restrict__ counts) {
    const int e = threadIdx.x;
    if (e >= E_EXP) return;
    int pos = 0;
    for (int a = 0; a < T_TOK * K_TOP; ++a) {
        if (eidx_flat[a] == e) {
            pos_flat[a] = pos;
            if (pos < CAP) tok_list[e * CAP + pos] = a / K_TOP;
            ++pos;
        }
    }
    counts[e] = pos < CAP ? pos : CAP;
}

// ------------------------- GEMM1: X @ Wgu -> silu*u ------------------------
// Block = 8 waves. Tile: 32 tokens x 64 act-cols. Waves 0-3 gate columns,
// waves 4-7 matching up columns; each wave does 2 WMMA (M=0..15 / 16..31)
// sharing one B fragment. A staged through LDS (ISA 16-bit A 16x32 layout).
__global__ __launch_bounds__(256)
void moe_gemm1_silu_kernel(const float* __restrict__ X,      // [*, Kdim]
                           const float* __restrict__ W,      // [Kdim, 2*Nact] per expert
                           size_t w_estride,
                           int Nact,
                           const int* __restrict__ tok_list, // [E,CAP] or null
                           const int* __restrict__ counts,   // [E] or null
                           int count_const,
                           __bf16* __restrict__ act,         // [CAP,Nact] per expert
                           size_t act_estride,
                           int Kdim) {
    const int e     = blockIdx.z;
    const int count = counts ? min(counts[e], CAP) : count_const;
    const int m0    = blockIdx.y * 32;
    if (m0 >= count) return;
    const int n0 = blockIdx.x * 64;

    const float* We   = W + (size_t)e * w_estride;
    __bf16*      acte = act + (size_t)e * act_estride;
    const int*   tl   = tok_list ? tok_list + e * CAP : nullptr;
    const int    ldW  = 2 * Nact;

    __shared__ __align__(16) __bf16 As[32][32];
    __shared__ float Us[4][32][16];

    const int tid  = threadIdx.x;
    const int wv   = tid >> 5;
    const int lane = tid & 31;
    const int lhi  = lane >> 4;
    const int llo  = lane & 15;
    const int colw = (wv & 3) * 16;
    const int col  = n0 + colw + ((wv >= 4) ? Nact : 0) + llo;

    // --- loop-invariant A staging setup: thread -> (row sr, 4 k-values) ---
    const int sr = tid >> 3;            // 0..31
    const int sk = (tid & 7) * 4;       // 0..28
    const int sm = m0 + sr;
    const bool valid = sm < count;
    const int  safem = valid ? sm : m0;             // m0 < count always
    const int  grow  = tl ? tl[safem] : safem;
    const float ascale = valid ? 1.f : 0.f;
    const float* __restrict__ srcp = X + (size_t)grow * Kdim + sk;

    // --- loop-invariant B pointer ---
    const float* __restrict__ wp = We + (size_t)(lhi * 8) * ldW + col;
    const size_t wstep = (size_t)32 * ldW;

    v8f c0 = {}, c1 = {};
    for (int kc = 0; kc < Kdim; kc += 32) {
        // cooperative A stage: 32 rows x 32 K, fp32 -> bf16, branch-free
        const float4 xv = *(const float4*)(srcp + kc);
        v4bf pk;
        pk[0] = (__bf16)(xv.x * ascale);
        pk[1] = (__bf16)(xv.y * ascale);
        pk[2] = (__bf16)(xv.z * ascale);
        pk[3] = (__bf16)(xv.w * ascale);
        *(v4bf*)&As[sr][sk] = pk;
        __syncthreads();

        // A fragments per ISA layout: two 16B LDS reads each
        const v8bf a0lo = *(const v8bf*)&As[llo][lhi * 8];
        const v8bf a0hi = *(const v8bf*)&As[llo][16 + lhi * 8];
        const v8bf a1lo = *(const v8bf*)&As[16 + llo][lhi * 8];
        const v8bf a1hi = *(const v8bf*)&As[16 + llo][16 + lhi * 8];
        v16bf a0, a1;
        #pragma unroll
        for (int j = 0; j < 8; ++j) {
            a0[j] = a0lo[j]; a0[8 + j] = a0hi[j];
            a1[j] = a1lo[j]; a1[8 + j] = a1hi[j];
        }
        // B fragment: element j = W[kc + lhi*8 + (j<8 ? j : j+8)][col], fp32 -> bf16
        v16bf b;
        #pragma unroll
        for (int j = 0; j < 16; ++j)
            b[j] = (__bf16)wp[(size_t)(j < 8 ? j : j + 8) * ldW];
        __builtin_prefetch(wp + wstep, 0, 0);       // next k-block (speculative)

        c0 = __builtin_amdgcn_wmma_f32_16x16x32_bf16(false, a0, false, b,
                                                     (short)0, c0, false, false);
        c1 = __builtin_amdgcn_wmma_f32_16x16x32_bf16(false, a1, false, b,
                                                     (short)0, c1, false, false);
        wp += wstep;
        __syncthreads();
    }

    // pair gate/up tiles through LDS, apply silu(g)*u, store bf16 activations
    if (wv >= 4) {
        #pragma unroll
        for (int vv = 0; vv < 8; ++vv) {
            Us[wv - 4][vv + lhi * 8][llo]      = c0[vv];
            Us[wv - 4][16 + vv + lhi * 8][llo] = c1[vv];
        }
    }
    __syncthreads();
    if (wv < 4) {
        #pragma unroll
        for (int vv = 0; vv < 8; ++vv) {
            const int r0 = vv + lhi * 8;
            const int r1 = 16 + r0;
            const float g0 = c0[vv], u0 = Us[wv][r0][llo];
            const float g1 = c1[vv], u1 = Us[wv][r1][llo];
            const float av0 = (g0 / (1.f + expf(-g0))) * u0;
            const float av1 = (g1 / (1.f + expf(-g1))) * u1;
            acte[(size_t)(m0 + r0) * Nact + n0 + colw + llo] = (__bf16)av0;
            acte[(size_t)(m0 + r1) * Nact + n0 + colw + llo] = (__bf16)av1;
        }
    }
}

// ------------------------- GEMM2: act @ Wdown -> Y -------------------------
// Tile: 32 rows x 128 cols per block (8 waves x 16 cols, 2 WMMA per wave
// sharing one B fragment). A already bf16 in WMMA-A row layout.
__global__ __launch_bounds__(256)
void moe_gemm2_kernel(const __bf16* __restrict__ A,   // [CAP,Kdim] per expert
                      size_t a_estride,
                      const float* __restrict__ W,    // [Kdim,Nout] per expert
                      size_t w_estride,
                      float* __restrict__ Y,          // [CAP,Nout] per expert
                      size_t y_estride,
                      int Nout, int Kdim,
                      const int* __restrict__ counts, int count_const) {
    const int e     = blockIdx.z;
    const int count = counts ? min(counts[e], CAP) : count_const;
    const int m0    = blockIdx.y * 32;
    if (m0 >= count) return;
    const int tid  = threadIdx.x;
    const int wv   = tid >> 5;
    const int lane = tid & 31;
    const int lhi  = lane >> 4;
    const int llo  = lane & 15;
    const int col  = blockIdx.x * 128 + wv * 16 + llo;

    const __bf16* Ae = A + (size_t)e * a_estride;
    const float*  We = W + (size_t)e * w_estride;
    float*        Ye = Y + (size_t)e * y_estride;

    // loop-invariant A row pointers (rows m0+llo and m0+16+llo)
    const __bf16* __restrict__ a0p = Ae + (size_t)(m0 + llo) * Kdim + lhi * 8;
    const __bf16* __restrict__ a1p = Ae + (size_t)(m0 + 16 + llo) * Kdim + lhi * 8;
    const float*  __restrict__ wp  = We + (size_t)(lhi * 8) * Nout + col;
    const size_t wstep = (size_t)32 * Nout;

    v8f c0 = {}, c1 = {};
    for (int kc = 0; kc < Kdim; kc += 32) {
        const v8bf a0lo = *(const v8bf*)(a0p + kc);
        const v8bf a0hi = *(const v8bf*)(a0p + kc + 16);
        const v8bf a1lo = *(const v8bf*)(a1p + kc);
        const v8bf a1hi = *(const v8bf*)(a1p + kc + 16);
        v16bf a0, a1;
        #pragma unroll
        for (int j = 0; j < 8; ++j) {
            a0[j] = a0lo[j]; a0[8 + j] = a0hi[j];
            a1[j] = a1lo[j]; a1[8 + j] = a1hi[j];
        }
        v16bf b;
        #pragma unroll
        for (int j = 0; j < 16; ++j)
            b[j] = (__bf16)wp[(size_t)(j < 8 ? j : j + 8) * Nout];
        __builtin_prefetch(wp + wstep, 0, 0);

        c0 = __builtin_amdgcn_wmma_f32_16x16x32_bf16(false, a0, false, b,
                                                     (short)0, c0, false, false);
        c1 = __builtin_amdgcn_wmma_f32_16x16x32_bf16(false, a1, false, b,
                                                     (short)0, c1, false, false);
        wp += wstep;
    }
    #pragma unroll
    for (int vv = 0; vv < 8; ++vv) {
        Ye[(size_t)(m0 + vv + lhi * 8) * Nout + col]      = c0[vv];
        Ye[(size_t)(m0 + 16 + vv + lhi * 8) * Nout + col] = c1[vv];
    }
}

// ------------------------------ Combine ------------------------------------
__global__ void moe_combine_kernel(const float* __restrict__ shared_out,
                                   const float* __restrict__ ye,
                                   const int* __restrict__ eidx_flat,
                                   const int* __restrict__ pos_flat,
                                   const float* __restrict__ w_flat,
                                   float* __restrict__ out) {
    const int t = blockIdx.y;
    const int h = blockIdx.x * 256 + threadIdx.x;
    float acc = shared_out[(size_t)t * H_DIM + h];
    #pragma unroll
    for (int k = 0; k < K_TOP; ++k) {
        const int a = t * K_TOP + k;
        const int e = eidx_flat[a];
        const int p = pos_flat[a];
        const float w = w_flat[a];
        if (p < CAP)
            acc += SCALE_F * w * ye[((size_t)e * CAP + p) * H_DIM + h];
    }
    out[(size_t)t * H_DIM + h] = acc;
}

// ------------------------------ Launcher -----------------------------------
extern "C" void kernel_launch(void* const* d_in, const int* in_sizes, int n_in,
                              void* d_out, int out_size, void* d_ws, size_t ws_size,
                              hipStream_t stream) {
    (void)in_sizes; (void)n_in; (void)out_size; (void)ws_size;
    const float* x      = (const float*)d_in[0];
    const float* gate_w = (const float*)d_in[1];
    const float* e_bias = (const float*)d_in[2];
    const float* w_gu   = (const float*)d_in[3];   // [E, H, 2I]
    const float* w_dn   = (const float*)d_in[4];   // [E, I, H]
    const float* ws_gu  = (const float*)d_in[5];   // [H, 2*ISH]
    const float* ws_dn  = (const float*)d_in[6];   // [ISH, H]
    float* out = (float*)d_out;

    char* ws = (char*)d_ws;
    size_t off = 0;
    auto carve = [&](size_t bytes) -> void* {
        void* p = ws + off;
        off = (off + bytes + 255) & ~(size_t)255;
        return p;
    };
    int*    counts    = (int*)carve(E_EXP * 4);
    int*    tok_list  = (int*)carve((size_t)E_EXP * CAP * 4);
    int*    eidx_flat = (int*)carve((size_t)T_TOK * K_TOP * 4);
    int*    pos_flat  = (int*)carve((size_t)T_TOK * K_TOP * 4);
    float*  w_flat    = (float*)carve((size_t)T_TOK * K_TOP * 4);
    __bf16* act_r     = (__bf16*)carve((size_t)E_EXP * CAP * I_DIM * 2);
    __bf16* act_s     = (__bf16*)carve((size_t)T_TOK * ISH * 2);
    float*  ye        = (float*)carve((size_t)E_EXP * CAP * H_DIM * 4);
    float*  sh_out    = (float*)carve((size_t)T_TOK * H_DIM * 4);

    moe_router_kernel<<<T_TOK, 256, 0, stream>>>(x, gate_w, e_bias, eidx_flat, w_flat);
    moe_dispatch_kernel<<<1, 64, 0, stream>>>(eidx_flat, pos_flat, tok_list, counts);

    // routed up-projection + silu*mul: act_r[e, slot, :I]
    dim3 g1r(I_DIM / 64, CAP / 32, E_EXP);
    moe_gemm1_silu_kernel<<<g1r, 256, 0, stream>>>(
        x, w_gu, (size_t)H_DIM * 2 * I_DIM, I_DIM,
        tok_list, counts, 0, act_r, (size_t)CAP * I_DIM, H_DIM);

    // shared-expert up-projection + silu*mul: act_s[t, :ISH]
    dim3 g1s(ISH / 64, T_TOK / 32, 1);
    moe_gemm1_silu_kernel<<<g1s, 256, 0, stream>>>(
        x, ws_gu, 0, ISH,
        nullptr, nullptr, T_TOK, act_s, 0, H_DIM);

    // routed down-projection: ye[e, slot, :H]
    dim3 g2r(H_DIM / 128, CAP / 32, E_EXP);
    moe_gemm2_kernel<<<g2r, 256, 0, stream>>>(
        act_r, (size_t)CAP * I_DIM, w_dn, (size_t)I_DIM * H_DIM,
        ye, (size_t)CAP * H_DIM, H_DIM, I_DIM, counts, 0);

    // shared down-projection: sh_out[t, :H]
    dim3 g2s(H_DIM / 128, T_TOK / 32, 1);
    moe_gemm2_kernel<<<g2s, 256, 0, stream>>>(
        act_s, 0, ws_dn, 0, sh_out, 0, H_DIM, ISH, nullptr, T_TOK);

    // gather/scale combine
    dim3 gc(H_DIM / 256, T_TOK);
    moe_combine_kernel<<<gc, 256, 0, stream>>>(sh_out, ye, eidx_flat, pos_flat,
                                               w_flat, out);
}
